// SelectiveSSM_10024453669655
// MI455X (gfx1250) — matrix operand order repaired and checked by reading, not compiled, not run on hardware
//
#include <hip/hip_runtime.h>
#include <math.h>

#define DM   1024
#define DSZ  16
#define BSZ  8
#define SEQ  2048

#define BM 128
#define BN 128
#define BK 32
#define NKSTEPS (DM / BK)     // 32
#define LDS_STRIDE 40         // 32 + 8 pad (bf16 elems); row pitch 80B, 16B-aligned chunks

typedef __attribute__((ext_vector_type(16))) __bf16 v16bf;
typedef __attribute__((ext_vector_type(8)))  float  v8f;
typedef __attribute__((ext_vector_type(4)))  int    v4i;

union FragBF {
  v16bf v;
  uint4 q[2];
};

// ---- CDNA5 async global->LDS copy path (ASYNCcnt), with safe fallback ----
#if __has_builtin(__builtin_amdgcn_global_load_async_to_lds_b128)
#define HAVE_ASYNC_LDS 1
#endif

// builtin signature: (v4i addrspace(1)*, v4i addrspace(3)*, imm offset, imm cpol)
typedef __attribute__((address_space(1))) v4i as1_v4i;
typedef __attribute__((address_space(3))) v4i as3_v4i;

__device__ __forceinline__ void copy16_to_lds(const __bf16* g, __bf16* l) {
#ifdef HAVE_ASYNC_LDS
  __builtin_amdgcn_global_load_async_to_lds_b128(
      (as1_v4i*)(void*)g, (as3_v4i*)(void*)l, 0, 0);
#else
  *(uint4*)l = *(const uint4*)g;
#endif
}

#ifdef HAVE_ASYNC_LDS
#if __has_builtin(__builtin_amdgcn_s_wait_asynccnt)
#define WAIT_ASYNC(n) __builtin_amdgcn_s_wait_asynccnt(n)
#else
#define WAIT_ASYNC(n) asm volatile("s_wait_asynccnt %0" ::"i"(n))
#endif
#else
#define WAIT_ASYNC(n) ((void)0)
#endif

// ---------------- 0) one-shot dt_w f32 -> bf16 (2 MB) ----------------
__global__ __launch_bounds__(256)
void ssm_cvtw_kernel(const float* __restrict__ w, __bf16* __restrict__ wb) {
  size_t i = ((size_t)blockIdx.x * blockDim.x + threadIdx.x) * 4;
  float4 v = *(const float4*)(w + i);
  union { __bf16 h[4]; uint2 u; } pk;
  pk.h[0] = (__bf16)v.x; pk.h[1] = (__bf16)v.y;
  pk.h[2] = (__bf16)v.z; pk.h[3] = (__bf16)v.w;
  *(uint2*)(wb + i) = pk.u;
}

// ---------------- 1) depthwise conv k=3, pad=1 -> bf16 xf ----------------
__global__ __launch_bounds__(256)
void ssm_conv_kernel(const float* __restrict__ x,
                     const float* __restrict__ cw,   // (DM,1,3)
                     const float* __restrict__ cb,   // (DM,)
                     __bf16* __restrict__ xfb) {
  size_t idx = ((size_t)blockIdx.x * blockDim.x + threadIdx.x) * 2;
  int c = (int)(idx % DM);                 // even; c and c+1 share t
  int t = (int)((idx / DM) % SEQ);
  float2 mid = *(const float2*)(x + idx);
  float2 left  = make_float2(0.f, 0.f);
  float2 right = make_float2(0.f, 0.f);
  if (t > 0)       left  = *(const float2*)(x + idx - DM);
  if (t < SEQ - 1) right = *(const float2*)(x + idx + DM);
  float r0 = cw[c * 3 + 0] * left.x + cw[c * 3 + 1] * mid.x + cw[c * 3 + 2] * right.x + cb[c];
  float r1 = cw[(c + 1) * 3 + 0] * left.y + cw[(c + 1) * 3 + 1] * mid.y +
             cw[(c + 1) * 3 + 2] * right.y + cb[c + 1];
  union { __bf16 h[2]; unsigned u; } pk;
  pk.h[0] = (__bf16)r0; pk.h[1] = (__bf16)r1;
  *(unsigned*)(xfb + idx) = pk.u;
}

// ---------------- 2) dt GEMM: bf16 WMMA, async-staged, double-buffered ----------
// dt[m,n] = sigmoid( sum_k xf[m,k]*dt_w[n,k] + dt_b[n] ) * 0.099 + 0.001
__global__ __launch_bounds__(256)
void ssm_dt_gemm_kernel(const __bf16* __restrict__ Ag,   // xf bf16, (16384 x 1024)
                        const __bf16* __restrict__ Bg,   // dt_w bf16, (1024 x 1024)
                        const float* __restrict__ bias,  // dt_b
                        float* __restrict__ dtOut) {
  __shared__ __attribute__((aligned(16))) __bf16 As[2][BM * LDS_STRIDE];
  __shared__ __attribute__((aligned(16))) __bf16 Bs[2][BN * LDS_STRIDE];

  const int tid   = threadIdx.x;
  const int lane  = tid & 31;
  const int wv    = tid >> 5;
  const int waveM = wv >> 2;          // 0..1  -> 64 rows each
  const int waveN = wv & 3;           // 0..3  -> 32 cols each
  const int half  = lane >> 4;
  const int l15   = lane & 15;

  const int mBase = blockIdx.x * BM;
  const int nBase = blockIdx.y * BN;

  // staging: each thread moves 2 16B chunks of A and 2 of B per K-step
  const int r0 = tid >> 2;            // 0..63
  const int cg = (tid & 3) * 8;       // bf16 elem offset in 32-elem row

  auto stage = [&](int k0, int buf) {
    copy16_to_lds(Ag + (size_t)(mBase + r0) * DM + k0 + cg,
                  &As[buf][r0 * LDS_STRIDE + cg]);
    copy16_to_lds(Ag + (size_t)(mBase + 64 + r0) * DM + k0 + cg,
                  &As[buf][(64 + r0) * LDS_STRIDE + cg]);
    copy16_to_lds(Bg + (size_t)(nBase + r0) * DM + k0 + cg,
                  &Bs[buf][r0 * LDS_STRIDE + cg]);
    copy16_to_lds(Bg + (size_t)(nBase + 64 + r0) * DM + k0 + cg,
                  &Bs[buf][(64 + r0) * LDS_STRIDE + cg]);
  };

  v8f acc[4][2];
  #pragma unroll
  for (int i = 0; i < 4; ++i)
    #pragma unroll
    for (int j = 0; j < 2; ++j)
      #pragma unroll
      for (int r = 0; r < 8; ++r)
        acc[i][j][r] = 0.f;

  stage(0, 0);   // prologue: 4 async copies outstanding

  for (int kt = 0; kt < NKSTEPS; ++kt) {
    const int cur = kt & 1;
    if (kt + 1 < NKSTEPS) {
      stage((kt + 1) * BK, cur ^ 1);  // prefetch next tile into idle buffer
      WAIT_ASYNC(4);                  // retire this wave's copies for `cur`
    } else {
      WAIT_ASYNC(0);
    }
    __syncthreads();                  // cross-wave visibility of `cur`

    // A fragment (16x32 bf16): lanes 0-15 K {0..7,16..23}; lanes 16-31 K {8..15,24..31}
    FragBF a[4];
    #pragma unroll
    for (int i = 0; i < 4; ++i) {
      const __bf16* src = &As[cur][(waveM * 64 + i * 16 + l15) * LDS_STRIDE];
      a[i].q[0] = *(const uint4*)(src + half * 8);
      a[i].q[1] = *(const uint4*)(src + 16 + half * 8);
    }
    // B fragment (32x16 bf16): lane = column; lanes 0-15 K {0..7,16..23}; lanes 16-31 K {8..15,24..31}
    FragBF bfr[2];
    #pragma unroll
    for (int j = 0; j < 2; ++j) {
      const __bf16* src = &Bs[cur][(waveN * 32 + j * 16 + l15) * LDS_STRIDE];
      bfr[j].q[0] = *(const uint4*)(src + half * 8);
      bfr[j].q[1] = *(const uint4*)(src + 16 + half * 8);
    }

    #pragma unroll
    for (int i = 0; i < 4; ++i)
      #pragma unroll
      for (int j = 0; j < 2; ++j)
        acc[i][j] = __builtin_amdgcn_wmma_f32_16x16x32_bf16(
            false, a[i].v, false, bfr[j].v, (short)0, acc[i][j], false, false);

    __syncthreads();                  // all waves done reading `cur` before overwrite
  }

  // epilogue: bias + sigmoid scaling; C/D layout: n = lane&15, m = r + 8*(lane>>4)
  #pragma unroll
  for (int j = 0; j < 2; ++j) {
    int n = nBase + waveN * 32 + j * 16 + l15;
    float bval = bias[n];
    #pragma unroll
    for (int i = 0; i < 4; ++i) {
      #pragma unroll
      for (int r = 0; r < 8; ++r) {
        int m = mBase + waveM * 64 + i * 16 + r + 8 * half;
        float v = acc[i][j][r] + bval;
        float sg = 1.f / (1.f + __expf(-v));
        dtOut[(size_t)m * DM + n] = sg * 0.099f + 0.001f;
      }
    }
  }
}

// ---------------- 3) selective scan: 8192 independent recurrences ----------------
__global__ __launch_bounds__(256)
void ssm_scan_kernel(const __bf16* __restrict__ xfb,
                     const float* __restrict__ dt,
                     const float* __restrict__ a_log,
                     const float* __restrict__ bmat,
                     const float* __restrict__ cmat,
                     float* __restrict__ y) {
  int gid = blockIdx.x * blockDim.x + threadIdx.x;   // 0..8191
  int c = gid % DM;
  int b = gid / DM;

  float A[DSZ], Bc[DSZ], Cc[DSZ], s[DSZ];
  #pragma unroll
  for (int d = 0; d < DSZ; ++d) {
    float al = a_log[c * DSZ + d];
    float sp = (al > 20.f) ? al : log1pf(__expf(al));  // softplus
    A[d]  = -sp;
    Bc[d] = bmat[c * DSZ + d];
    Cc[d] = cmat[c * DSZ + d];
    s[d]  = 0.f;
  }

  size_t base = (size_t)b * SEQ * DM + c;
  for (int t = 0; t < SEQ; ++t) {
    size_t off = base + (size_t)t * DM;
    if (t + 16 < SEQ) {
      __builtin_prefetch(xfb + off + 16 * DM, 0, 0);   // global_prefetch_b8
      __builtin_prefetch(dt + off + 16 * DM, 0, 0);
    }
    float xt  = (float)xfb[off];
    float dtt = dt[off];
    float u   = dtt * xt;
    float yt  = 0.f;
    #pragma unroll
    for (int d = 0; d < DSZ; ++d) {
      s[d] = __expf(dtt * A[d]) * s[d] + u * Bc[d];
      yt  += s[d] * Cc[d];
    }
    y[off] = yt;
  }
}

extern "C" void kernel_launch(void* const* d_in, const int* in_sizes, int n_in,
                              void* d_out, int out_size, void* d_ws, size_t ws_size,
                              hipStream_t stream) {
  (void)in_sizes; (void)n_in; (void)out_size; (void)ws_size;
  const float* x      = (const float*)d_in[0];
  const float* a_log  = (const float*)d_in[1];
  const float* bmat   = (const float*)d_in[2];
  const float* cmat   = (const float*)d_in[3];
  const float* dt_w   = (const float*)d_in[4];
  const float* dt_b   = (const float*)d_in[5];
  const float* conv_w = (const float*)d_in[6];
  const float* conv_b = (const float*)d_in[7];
  float* out = (float*)d_out;

  const size_t elems = (size_t)BSZ * SEQ * DM;       // 16.7M
  // ws layout: dt f32 (64MB) | xf bf16 (32MB) | dt_w bf16 (2MB)  => 98MB total
  float*  dt  = (float*)d_ws;
  __bf16* xfb = (__bf16*)(dt + elems);
  __bf16* wb  = xfb + elems;

  ssm_cvtw_kernel<<<(DM * DM / 4) / 256, 256, 0, stream>>>(dt_w, wb);
  ssm_conv_kernel<<<(int)(elems / 2 / 256), 256, 0, stream>>>(x, conv_w, conv_b, xfb);

  dim3 g(BSZ * SEQ / BM, DM / BN);                   // (128, 8)
  ssm_dt_gemm_kernel<<<g, 256, 0, stream>>>(xfb, wb, dt_b, dt);

  ssm_scan_kernel<<<(BSZ * DM) / 256, 256, 0, stream>>>(xfb, dt, a_log, bmat, cmat, out);
}
